// ClosingPricePredictor_67147518706114
// MI455X (gfx1250) — matrix operand order repaired and checked by reading, not compiled, not run on hardware
//
#include <hip/hip_runtime.h>
#include <hip/hip_bf16.h>
#include <math.h>

#define NB   512      // batch
#define LSEQ 512      // timesteps
#define ID   64       // input size
#define HD   1024     // hidden size
#define FH   4096     // 4*H (gate dim)
#define KT   1088     // I + H (combined K)

typedef __bf16 bf16_t;
typedef __bf16 v16bf  __attribute__((ext_vector_type(16)));
typedef __bf16 bf16x8 __attribute__((ext_vector_type(8)));
typedef float  v8f    __attribute__((ext_vector_type(8)));

// ---------------- one-time conversion kernels ----------------
__global__ void f32_to_bf16_kernel(const float* __restrict__ src,
                                   bf16_t* __restrict__ dst, int n) {
  int i = blockIdx.x * blockDim.x + threadIdx.x;
  int stride = gridDim.x * blockDim.x;
  for (; i < n; i += stride) dst[i] = (bf16_t)src[i];
}

__global__ void copy_f32_kernel(const float* __restrict__ src,
                                float* __restrict__ dst, int n) {
  int i = blockIdx.x * blockDim.x + threadIdx.x;
  int stride = gridDim.x * blockDim.x;
  for (; i < n; i += stride) dst[i] = src[i];
}

// Build WT[n][k] = bf16 of (k<64 ? Wx[k][n] : Wh[k-64][n]); n in [0,4H), k in [0,1088).
// Transposed+concatenated so B tiles are contiguous-in-K for 16B async copies.
__global__ void build_wt_kernel(const float* __restrict__ Wx,
                                const float* __restrict__ Wh,
                                bf16_t* __restrict__ WT) {
  size_t i = (size_t)blockIdx.x * blockDim.x + threadIdx.x;
  size_t total = (size_t)FH * KT;
  size_t stride = (size_t)gridDim.x * blockDim.x;
  for (; i < total; i += stride) {
    int n = (int)(i / KT);
    int k = (int)(i % KT);
    float v = (k < ID) ? Wx[(size_t)k * FH + n] : Wh[(size_t)(k - ID) * FH + n];
    WT[i] = (bf16_t)v;
  }
}

// ---------------- fused LSTM step ----------------
// Block: 256 threads = 8 waves. Tile: 64 rows x 32 hidden cols x 4 gates.
// Wave w: gate = w&3, row-half = w>>2. 4 WMMA 16x16x32 bf16 tiles per wave.
// Triple-buffered LDS staging filled by GLOBAL_LOAD_ASYNC_TO_LDS_B128;
// one workgroup barrier per K-chunk.
struct SharedU {
  union {
    struct {
      alignas(16) bf16_t As[3][64 * 40];    // [buf][m][k], stride 40 bf16 = 80B
      alignas(16) bf16_t Bs[3][128 * 40];   // [buf][r][k], r = gate*32 + n_local
    } stage;                                // 46080 B
    float gbuf[4][64][33];                  // 33792 B fp32 gate pre-activations
  };
};

__device__ __forceinline__ void async_b128(uint32_t lds_addr, const void* gaddr) {
  asm volatile("global_load_async_to_lds_b128 %0, %1, off"
               :: "v"(lds_addr), "v"(gaddr) : "memory");
}

__device__ __forceinline__ v16bf ld_frag16(const bf16_t* p0, const bf16_t* p1) {
  // two 16B-aligned b128 LDS loads merged into one 16-element fragment
  bf16x8 lo = *reinterpret_cast<const bf16x8*>(p0);
  bf16x8 hi = *reinterpret_cast<const bf16x8*>(p1);
  return __builtin_shufflevector(lo, hi, 0, 1, 2, 3, 4, 5, 6, 7,
                                         8, 9, 10, 11, 12, 13, 14, 15);
}

__launch_bounds__(256)
__global__ void lstm_step_kernel(const bf16_t* __restrict__ xb,   // (N,L,I) bf16
                                 const bf16_t* __restrict__ WT,   // (4H,KT) bf16
                                 const float*  __restrict__ bias, // (4H)    f32
                                 const bf16_t* __restrict__ h_in, // (N,H)   bf16
                                 bf16_t* __restrict__ h_out,      // (N,H)   bf16
                                 float*  __restrict__ h_out_f,    // (N,H)   f32
                                 float*  __restrict__ c,          // (N,H)   f32 in/out
                                 int t) {
  __shared__ SharedU lds;

  const int tid  = threadIdx.x;
  const int lane = tid & 31;
  const int wave = tid >> 5;
  const int gate = wave & 3;
  const int wr   = wave >> 2;            // 0/1 -> 32-row half

  const int rowBase = blockIdx.y * 64;   // over N
  const int colBase = blockIdx.x * 32;   // over H (per-gate column)

  const int laneM = lane & 15;
  const int kbA   = (lane < 16) ? 0 : 8;   // A frag K base (ISA 16-bit A layout)
  const int kbB   = (lane < 16) ? 0 : 8;   // B frag K base (same K order as A)

  // Per-thread fixed positions for the async staging copies.
  const int mA   = tid >> 2;             // A tile row this thread copies
  const int k16A = tid & 3;              // which 16B chunk within the row
  const bf16_t* xRow = xb + ((size_t)(rowBase + mA) * LSEQ + (size_t)t) * ID + k16A * 8;
  const bf16_t* hRow = h_in + (size_t)(rowBase + mA) * HD + k16A * 8;

  // Issue one K-chunk (32 wide) of async global->LDS copies: 3 B128 per thread.
  auto issue_chunk = [&](int kc, int buf) {
    // ---- A tile: 64 rows x 64B, 256 chunks, 1 per thread ----
    {
      const bf16_t* gA = (kc < 2) ? (xRow + kc * 32) : (hRow + (kc - 2) * 32);
      uint32_t la = (uint32_t)(uintptr_t)&lds.stage.As[buf][mA * 40 + k16A * 8];
      async_b128(la, gA);
    }
    // ---- B tiles: 128 rows (4 gates x 32 n) x 64B, 512 chunks, 2 per thread ----
#pragma unroll
    for (int e = 0; e < 2; ++e) {
      int linear = e * 256 + tid;
      int r   = linear >> 2;          // 0..127
      int k16 = linear & 3;
      int n_g = (r >> 5) * HD + colBase + (r & 31);   // gate-dim row of WT
      const bf16_t* gB = WT + (size_t)n_g * KT + kc * 32 + k16 * 8;
      uint32_t lb = (uint32_t)(uintptr_t)&lds.stage.Bs[buf][r * 40 + k16 * 8];
      async_b128(lb, gB);
    }
  };

  v8f acc[2][2];
#pragma unroll
  for (int i = 0; i < 2; ++i)
#pragma unroll
    for (int j = 0; j < 2; ++j)
      acc[i][j] = (v8f){0.f, 0.f, 0.f, 0.f, 0.f, 0.f, 0.f, 0.f};

  constexpr int KCH = KT / 32;  // 34 chunks; 0..1 from x@Wx, 2..33 from h@Wh
  issue_chunk(0, 0);

  for (int kc = 0; kc < KCH; ++kc) {
    const int cur = kc % 3;
    if (kc + 1 < KCH) {
      issue_chunk(kc + 1, (kc + 1) % 3);
      // chunk kc+1's 3 ops in flight; in-order completion => <=3 outstanding
      // means chunk kc's copies have landed in LDS.
      asm volatile("s_wait_asynccnt 0x3" ::: "memory");
    } else {
      asm volatile("s_wait_asynccnt 0x0" ::: "memory");
    }
    __syncthreads();   // single barrier per chunk (triple buffering)

    // fragments (CDNA5 wave32 16-bit A 16x32 / B 32x16 layouts), b128 LDS loads
    v16bf afrag[2], bfrag[2];
#pragma unroll
    for (int ti = 0; ti < 2; ++ti) {
      const bf16_t* rp = &lds.stage.As[cur][(wr * 32 + ti * 16 + laneM) * 40];
      afrag[ti] = ld_frag16(rp + kbA, rp + kbA + 16);
    }
#pragma unroll
    for (int tj = 0; tj < 2; ++tj) {
      const bf16_t* rp = &lds.stage.Bs[cur][(gate * 32 + tj * 16 + laneM) * 40];
      bfrag[tj] = ld_frag16(rp + kbB, rp + kbB + 16);
    }

#pragma unroll
    for (int ti = 0; ti < 2; ++ti)
#pragma unroll
      for (int tj = 0; tj < 2; ++tj)
        acc[ti][tj] = __builtin_amdgcn_wmma_f32_16x16x32_bf16(
            false, afrag[ti], false, bfrag[tj], (short)0, acc[ti][tj],
            false, false);
  }
  __syncthreads();  // all waves done with last chunk before gbuf aliases stage

  // dump accumulators into gbuf
#pragma unroll
  for (int ti = 0; ti < 2; ++ti)
#pragma unroll
    for (int tj = 0; tj < 2; ++tj) {
      int m0 = wr * 32 + ti * 16 + ((lane >> 4) << 3);
      int n0 = tj * 16 + laneM;
#pragma unroll
      for (int r = 0; r < 8; ++r) lds.gbuf[gate][m0 + r][n0] = acc[ti][tj][r];
    }
  __syncthreads();

  // fused gate activations + state update (fp32), 8 elems/thread
#pragma unroll
  for (int e = 0; e < 8; ++e) {
    int linear = e * 256 + tid;
    int m = linear >> 5;
    int n = linear & 31;
    int row = rowBase + m;
    int col = colBase + n;
    float gi = lds.gbuf[0][m][n] + bias[col];
    float gf = lds.gbuf[1][m][n] + bias[HD + col];
    float gg = lds.gbuf[2][m][n] + bias[2 * HD + col];
    float go = lds.gbuf[3][m][n] + bias[3 * HD + col];
    gi = 1.f / (1.f + __expf(-gi));
    gf = 1.f / (1.f + __expf(-gf));
    gg = tanhf(gg);
    go = 1.f / (1.f + __expf(-go));
    size_t idx = (size_t)row * HD + col;
    float cn = gf * c[idx] + gi * gg;
    float hn = go * tanhf(cn);
    c[idx]       = cn;
    h_out[idx]   = (bf16_t)hn;
    h_out_f[idx] = hn;
  }
}

// ---------------- MLP head (tiny; fp32 VALU) ----------------
__global__ void head1_kernel(const float* __restrict__ h, const float* __restrict__ W1,
                             const float* __restrict__ b1, float* __restrict__ z) {
  int idx = blockIdx.x * blockDim.x + threadIdx.x;  // N*H threads
  int row = idx >> 10;
  int col = idx & 1023;
  const float* hp = h + (size_t)row * HD;
  float s = b1[col];
  for (int k = 0; k < HD; ++k) s += hp[k] * W1[(size_t)k * HD + col];
  z[idx] = 1.f / (1.f + __expf(-s));
}

__global__ void head2_kernel(const float* __restrict__ z, const float* __restrict__ W2,
                             const float* __restrict__ b2, float* __restrict__ out) {
  __shared__ float red[256];
  int row = blockIdx.x;
  float s = 0.f;
  for (int k = threadIdx.x; k < HD; k += 256)
    s += z[(size_t)row * HD + k] * W2[k];
  red[threadIdx.x] = s;
  __syncthreads();
  for (int off = 128; off > 0; off >>= 1) {
    if (threadIdx.x < off) red[threadIdx.x] += red[threadIdx.x + off];
    __syncthreads();
  }
  if (threadIdx.x == 0) out[row] = red[0] + b2[0];
}

// ---------------- launch ----------------
extern "C" void kernel_launch(void* const* d_in, const int* in_sizes, int n_in,
                              void* d_out, int out_size, void* d_ws, size_t ws_size,
                              hipStream_t stream) {
  (void)in_sizes; (void)n_in; (void)out_size; (void)ws_size;
  const float* x  = (const float*)d_in[0];
  const float* c0 = (const float*)d_in[1];
  const float* h0 = (const float*)d_in[2];
  const float* Wx = (const float*)d_in[3];
  const float* Wh = (const float*)d_in[4];
  const float* b  = (const float*)d_in[5];
  const float* W1 = (const float*)d_in[6];
  const float* b1 = (const float*)d_in[7];
  const float* W2 = (const float*)d_in[8];
  const float* b2 = (const float*)d_in[9];
  float* out = (float*)d_out;

  // workspace carve (~51 MB)
  char* ws = (char*)d_ws;
  size_t off = 0;
  auto carve = [&](size_t bytes) {
    char* p = ws + off;
    off = (off + bytes + 255) & ~(size_t)255;
    return p;
  };
  bf16_t* xb   = (bf16_t*)carve((size_t)NB * LSEQ * ID * 2);
  bf16_t* WT   = (bf16_t*)carve((size_t)FH * KT * 2);
  bf16_t* hA   = (bf16_t*)carve((size_t)NB * HD * 2);
  bf16_t* hB   = (bf16_t*)carve((size_t)NB * HD * 2);
  float*  cbuf = (float*)carve((size_t)NB * HD * 4);
  float*  hf   = (float*)carve((size_t)NB * HD * 4);
  float*  z    = (float*)carve((size_t)NB * HD * 4);

  // one-time conversions (inputs never mutated)
  f32_to_bf16_kernel<<<2048, 256, 0, stream>>>(x,  xb, NB * LSEQ * ID);
  build_wt_kernel   <<<4096, 256, 0, stream>>>(Wx, Wh, WT);
  f32_to_bf16_kernel<<<256,  256, 0, stream>>>(h0, hA, NB * HD);
  copy_f32_kernel   <<<256,  256, 0, stream>>>(c0, cbuf, NB * HD);

  // 512 serial LSTM steps, h ping-pongs
  dim3 grid(HD / 32, NB / 64);
  for (int t = 0; t < LSEQ; ++t) {
    const bf16_t* hin  = (t & 1) ? hB : hA;
    bf16_t*       hout = (t & 1) ? hA : hB;
    lstm_step_kernel<<<grid, 256, 0, stream>>>(xb, WT, b, hin, hout, hf, cbuf, t);
  }

  // MLP head on fp32 h
  head1_kernel<<<NB * HD / 256, 256, 0, stream>>>(hf, W1, b1, z);
  head2_kernel<<<NB, 256, 0, stream>>>(z, W2, b2, out);
}
